// DeepAR_29764123361451
// MI455X (gfx1250) — 8-bit matrix operand order repaired and checked by reading, compiled, not run on hardware
//
#include <hip/hip_runtime.h>
#include <math.h>

typedef __attribute__((ext_vector_type(16))) int          v16i;
typedef __attribute__((ext_vector_type(8)))  float        v8f;
typedef __attribute__((ext_vector_type(4)))  unsigned int u4;

#define BATCH 1024
#define TT    512
#define EE    32
#define GATES 512
#define KPAD  128
#define INPF  101

#define L_W0    0
#define L_W1    65536
#define L_W2    131072
#define L_W3    196608
#define L_BIAS0 262144
#define L_BIAS1 264192
#define L_WMU   266240
#define L_WSIG  267264
#define L_GRP   268288
#define G_XA    0
#define G_H0A   4096
#define G_H1A   8192
#define G_MUA   12288
#define G_SGA   12352
#define GRP_BYTES 12416
#define GRP_SZ  12544
#define SMEM_TOTAL (L_GRP + 2*GRP_SZ)

__device__ inline unsigned char f32_to_fp8(float x) {
#if __has_builtin(__builtin_amdgcn_cvt_pk_fp8_f32)
  return (unsigned char)(__builtin_amdgcn_cvt_pk_fp8_f32(x, x, 0, false) & 0xff);
#else
  union { float f; unsigned u; } v; v.f = x;
  unsigned s = (v.u >> 24) & 0x80u;
  v.u &= 0x7fffffffu;
  if (v.u >= 0x7f800000u) return (unsigned char)(s | 0x7eu);
  if (v.f >= 448.0f)      return (unsigned char)(s | 0x7eu);
  if (v.f < 0.015625f) {
    int m = (int)(v.f * 512.0f + 0.5f);
    return (unsigned char)(s | (unsigned)m);
  }
  int exp = (int)(v.u >> 23) - 127;
  unsigned man = v.u & 0x7fffffu;
  unsigned r = (man + 0x7ffffu + ((man >> 20) & 1u)) >> 20;
  if (r == 8u) { r = 0u; exp += 1; }
  unsigned code = (unsigned)((exp + 7) << 3) | r;
  if (code >= 0x7fu) code = 0x7eu;
  return (unsigned char)(s | code);
#endif
}

__device__ inline v8f splat8(float x) { v8f r; for (int i = 0; i < 8; ++i) r[i] = x; return r; }
__device__ inline float sigm(float x)   { return 1.0f / (1.0f + __expf(-x)); }
__device__ inline float tanh_f(float x) { return 2.0f / (1.0f + __expf(-2.0f * x)) - 1.0f; }

__device__ inline v8f wmma128(v16i a, v16i b, v8f c) {
  return __builtin_amdgcn_wmma_f32_16x16x128_fp8_fp8(a, b, (short)0, c, false, false);
}

__device__ inline v16i load_a128(const unsigned char* abase, int m, int ksel) {
  const unsigned char* p = abase + m * 128 + ksel * 16;
  union { unsigned long long q[8]; v16i v; } u;
  u.q[0] = *(const unsigned long long*)(p + 0);
  u.q[1] = *(const unsigned long long*)(p + 8);
  u.q[2] = *(const unsigned long long*)(p + 32);
  u.q[3] = *(const unsigned long long*)(p + 40);
  u.q[4] = *(const unsigned long long*)(p + 64);
  u.q[5] = *(const unsigned long long*)(p + 72);
  u.q[6] = *(const unsigned long long*)(p + 96);
  u.q[7] = *(const unsigned long long*)(p + 104);
  return u.v;
}

__device__ inline v16i load_b128(const unsigned char* wbase, int nt, int l16, int ksel) {
  const unsigned char* p = wbase + (nt * 16 + l16) * 128 + ksel * 16;
  union { u4 o[4]; v16i v; } u;
  u.o[0] = *(const u4*)(p + 0);
  u.o[1] = *(const u4*)(p + 32);
  u.o[2] = *(const u4*)(p + 64);
  u.o[3] = *(const u4*)(p + 96);
  return u.v;
}

__device__ inline void lstm_layer(const unsigned char* abuf_x, const unsigned char* abuf_h,
                                  const unsigned char* wih, const unsigned char* whh,
                                  const float* bias, const float* wmu, const float* wsig,
                                  int layer, int w, int l16, int ksel,
                                  v8f& cst, v8f& hn, v8f& muP, v8f& sgP) {
  v16i ax = load_a128(abuf_x, l16, ksel);
  v16i ah = load_a128(abuf_h, l16, ksel);
  const int hid = 16 * w + l16;
  float bi = bias[hid], bf = bias[128 + hid], bg = bias[256 + hid], bo = bias[384 + hid];
  v8f acc[4];
  for (int gb = 0; gb < 4; ++gb) {
    acc[gb] = splat8(0.0f);
    acc[gb] = wmma128(ax, load_b128(wih, 8 * gb + w, l16, ksel), acc[gb]);
    acc[gb] = wmma128(ah, load_b128(whh, 8 * gb + w, l16, ksel), acc[gb]);
  }
  float wm = wmu[2 * hid + layer], ws_ = wsig[2 * hid + layer];
  for (int v = 0; v < 8; ++v) {
    float c_ = sigm(acc[1][v] + bf) * cst[v] + sigm(acc[0][v] + bi) * tanh_f(acc[2][v] + bg);
    float h_ = sigm(acc[3][v] + bo) * tanh_f(c_);
    cst[v] = c_;
    hn[v]  = h_;
    muP[v] += h_ * wm;
    sgP[v] += h_ * ws_;
  }
}

__device__ inline void store_h(unsigned char* hbuf, const v8f& hn, int w, int l16, int ksel) {
  int hid = 16 * w + l16;
  for (int v = 0; v < 8; ++v)
    hbuf[(v + 8 * ksel) * 128 + hid] = f32_to_fp8(hn[v]);
}

__device__ inline void build_x(unsigned char* xb, int gtid, int b0, int t,
                               const int* s_cat, const int* k_cat, const float* k_cont,
                               const float* emb_s, const float* emb_k0, const float* emb_k1) {
  for (int idx = gtid; idx < 2048; idx += 256) {
    int m = idx >> 7, k = idx & 127;
    if (k == 0 || k > 100) continue;
    int bt = (b0 + m) * TT + t;
    float val;
    if (k < 5)        val = k_cont[bt * 4 + (k - 1)];
    else if (k < 37)  val = emb_s [s_cat[bt]         * EE + (k - 5)];
    else if (k < 69)  val = emb_k0[k_cat[bt * 2 + 0] * EE + (k - 37)];
    else              val = emb_k1[k_cat[bt * 2 + 1] * EE + (k - 69)];
    xb[idx] = f32_to_fp8(val);
    if (k == 5 && t + 1 < TT) {
      int s2 = s_cat[bt + 1];
      __builtin_prefetch(emb_s + (size_t)s2 * EE, 0, 0);
      __builtin_prefetch(emb_s + (size_t)s2 * EE + 16, 0, 0);
    }
  }
}

extern "C" __global__ __launch_bounds__(512, 1)
void deepar_fused(const int* __restrict__ s_cat, const int* __restrict__ k_cat,
                  const float* __restrict__ k_cont, const float* __restrict__ target,
                  const float* __restrict__ emb_s, const float* __restrict__ emb_k0,
                  const float* __restrict__ emb_k1,
                  const float* __restrict__ W_ih0, const float* __restrict__ W_hh0,
                  const float* __restrict__ b_ih0, const float* __restrict__ b_hh0,
                  const float* __restrict__ W_ih1, const float* __restrict__ W_hh1,
                  const float* __restrict__ b_ih1, const float* __restrict__ b_hh1,
                  const float* __restrict__ W_mu, const float* __restrict__ b_mu,
                  const float* __restrict__ W_sig, const float* __restrict__ b_sig,
                  float* __restrict__ out) {
  extern __shared__ unsigned char smem[];
  const int tid   = threadIdx.x;
  const int lane  = tid & 31;
  const int l16   = lane & 15;
  const int ksel  = lane >> 4;
  const int wave  = tid >> 5;
  const int w     = wave & 7;
  const int group = wave >> 3;
  const int gtid  = tid & 255;
  const int b0    = blockIdx.x * 32 + group * 16;

  unsigned char* wih0 = smem + L_W0;
  unsigned char* whh0 = smem + L_W1;
  unsigned char* wih1 = smem + L_W2;
  unsigned char* whh1 = smem + L_W3;
  float* bias0 = (float*)(smem + L_BIAS0);
  float* bias1 = (float*)(smem + L_BIAS1);
  float* wmu   = (float*)(smem + L_WMU);
  float* wsig  = (float*)(smem + L_WSIG);
  unsigned char* grp = smem + L_GRP + group * GRP_SZ;
  float* mua = (float*)(grp + G_MUA);
  float* sga = (float*)(grp + G_SGA);

  for (int i = tid; i < GATES * KPAD; i += 512) {
    int n = i >> 7, k = i & 127;
    wih0[i] = (k < INPF) ? f32_to_fp8(W_ih0[n * INPF + k]) : (unsigned char)0;
    whh0[i] = f32_to_fp8(W_hh0[i]);
    wih1[i] = f32_to_fp8(W_ih1[i]);
    whh1[i] = f32_to_fp8(W_hh1[i]);
  }
  for (int i = tid; i < GATES; i += 512) {
    bias0[i] = b_ih0[i] + b_hh0[i];
    bias1[i] = b_ih1[i] + b_hh1[i];
  }
  for (int i = tid; i < 256; i += 512) { wmu[i] = W_mu[i]; wsig[i] = W_sig[i]; }
  for (int i = gtid; i < GRP_BYTES; i += 256) grp[i] = 0;
  const float bmu = b_mu[0], bsg = b_sig[0];
  __syncthreads();

  build_x(grp + G_XA, gtid, b0, 0, s_cat, k_cat, k_cont, emb_s, emb_k0, emb_k1);
  __syncthreads();

  v8f c0 = splat8(0.0f), c1 = splat8(0.0f);

  for (int t = 0; t < TT; ++t) {
    const int p = t & 1;
    unsigned char* xr  = grp + G_XA  + p * 2048;
    unsigned char* xw  = grp + G_XA  + (1 - p) * 2048;
    unsigned char* h0r = grp + G_H0A + p * 2048;
    unsigned char* h0w = grp + G_H0A + (1 - p) * 2048;
    unsigned char* h1r = grp + G_H1A + p * 2048;
    unsigned char* h1w = grp + G_H1A + (1 - p) * 2048;

    v8f muP = splat8(0.0f), sgP = splat8(0.0f);
    v8f h0n, h1n;

    lstm_layer(xr, h0r, wih0, whh0, bias0, wmu, wsig, 0, w, l16, ksel, c0, h0n, muP, sgP);
    store_h(h0w, h0n, w, l16, ksel);
    __syncthreads();

    lstm_layer(h0w, h1r, wih1, whh1, bias1, wmu, wsig, 1, w, l16, ksel, c1, h1n, muP, sgP);
    store_h(h1w, h1n, w, l16, ksel);
    if (t + 1 < TT)
      build_x(xw, gtid, b0, t + 1, s_cat, k_cat, k_cont, emb_s, emb_k0, emb_k1);

    for (int off = 1; off < 16; off <<= 1) {
      for (int v = 0; v < 8; ++v) {
        muP[v] += __shfl_xor(muP[v], off, 16);
        sgP[v] += __shfl_xor(sgP[v], off, 16);
      }
    }
    if (l16 == 0) {
      for (int v = 0; v < 8; ++v) {
        atomicAdd(&mua[v + 8 * ksel], muP[v]);
        atomicAdd(&sga[v + 8 * ksel], sgP[v]);
      }
    }
    __syncthreads();

    if (w == 0 && lane < 16) {
      int m = lane;
      float mu = mua[m] + bmu;
      float sp = sga[m] + bsg;
      float sig = (sp > 20.0f) ? sp : log1pf(__expf(sp));
      int bt = (b0 + m) * TT + t;
      float2 r; r.x = mu; r.y = sig;
      *(float2*)(out + (size_t)bt * 2) = r;
      mua[m] = 0.0f;
      sga[m] = 0.0f;
      if (t + 1 < TT) {
        float tr = target[bt];
        float x0 = (tr == 0.0f) ? mu : tr;
        xw[m * 128] = f32_to_fp8(x0);
      }
    }
    __syncthreads();
  }
}

extern "C" void kernel_launch(void* const* d_in, const int* in_sizes, int n_in,
                              void* d_out, int out_size, void* d_ws, size_t ws_size,
                              hipStream_t stream) {
  (void)in_sizes; (void)n_in; (void)d_ws; (void)ws_size; (void)out_size;
  const int*   s_cat  = (const int*)  d_in[0];
  const int*   k_cat  = (const int*)  d_in[1];
  const float* k_cont = (const float*)d_in[2];
  const float* target = (const float*)d_in[3];
  const float* emb_s  = (const float*)d_in[4];
  const float* emb_k0 = (const float*)d_in[5];
  const float* emb_k1 = (const float*)d_in[6];
  const float* W_ih0  = (const float*)d_in[7];
  const float* W_hh0  = (const float*)d_in[8];
  const float* b_ih0  = (const float*)d_in[9];
  const float* b_hh0  = (const float*)d_in[10];
  const float* W_ih1  = (const float*)d_in[11];
  const float* W_hh1  = (const float*)d_in[12];
  const float* b_ih1  = (const float*)d_in[13];
  const float* b_hh1  = (const float*)d_in[14];
  const float* W_mu   = (const float*)d_in[15];
  const float* b_mu   = (const float*)d_in[16];
  const float* W_sig  = (const float*)d_in[17];
  const float* b_sig  = (const float*)d_in[18];
  float* out = (float*)d_out;

  hipFuncSetAttribute(reinterpret_cast<const void*>(deepar_fused),
                      hipFuncAttributeMaxDynamicSharedMemorySize, (int)SMEM_TOTAL);
  deepar_fused<<<dim3(BATCH / 32), dim3(512), SMEM_TOTAL, stream>>>(
      s_cat, k_cat, k_cont, target, emb_s, emb_k0, emb_k1,
      W_ih0, W_hh0, b_ih0, b_hh0, W_ih1, W_hh1, b_ih1, b_hh1,
      W_mu, b_mu, W_sig, b_sig, out);
}
